// PointMixerIntraSetLayerPaperv3_9242769621752
// MI455X (gfx1250) — matrix operand order repaired and checked by reading, not compiled, not run on hardware
//
#include <hip/hip_runtime.h>
#include <hip/hip_bf16.h>
#include <math.h>

// ---------------- problem constants ----------------
#define NPTS  65536
#define CIN   128
#define OUTC  128
#define KNN   16
#define NKROW (NPTS*KNN)          // 1,048,576 rows
#define EPSV  1e-5f
#define CNTF  ((float)NKROW)

typedef _Float16 v16h __attribute__((ext_vector_type(16)));
typedef float    v8f  __attribute__((ext_vector_type(8)));

// ---------------- d_out layout (4-byte elements) ----------------
// return order: x_out [N,128], x_knn [N,K,128], knn_idx [N,K] (int bits), p_r [N,K,3]
static constexpr size_t DO_XOUT = 0;
static constexpr size_t DO_XKNN = (size_t)NPTS * OUTC;                       // 8,388,608
static constexpr size_t DO_IDX  = DO_XKNN + (size_t)NPTS * KNN * OUTC;       // 142,606,336
static constexpr size_t DO_PR   = DO_IDX  + (size_t)NPTS * KNN;              // 143,654,912

// ---------------- workspace layout (floats) ----------------
static constexpr size_t WS_EIN  = 0;                                          // [NK,32]
static constexpr size_t WS_BPRE = WS_EIN  + (size_t)NKROW * 32;               // [NK,16]
static constexpr size_t WS_Y    = WS_BPRE + (size_t)NKROW * 16;               // [N,128]
static constexpr size_t WS_PU   = WS_Y    + (size_t)NPTS * OUTC;              // [4096][6]
static constexpr size_t WS_PA   = WS_PU   + 4096 * 6;                         // [1024][256]
static constexpr size_t WS_PB   = WS_PA   + 1024 * 256;                       // [1024][32]
static constexpr size_t WS_BN1  = WS_PB   + 1024 * 32;                        // scale3+shift3
static constexpr size_t WS_BN2A = WS_BN1  + 8;                                // scale128+shift128
static constexpr size_t WS_BN2B = WS_BN2A + 256;                              // scale16+shift16

#define WMMA_F16(A,B,C) __builtin_amdgcn_wmma_f32_16x16x32_f16(false,(A),false,(B),(short)0,(C),false,false)

// =====================================================================
// K1a (light, thread-per-row): knn_idx copy, p_r compute+write,
// deterministic BN1 partial stats of u = p_r@Wp1+bp1.
// =====================================================================
__global__ void __launch_bounds__(256)
k1a_prep(const float* __restrict__ p, const int* __restrict__ knn,
         const float* __restrict__ Wp1, const float* __restrict__ bp1,
         float* __restrict__ out, float* __restrict__ ws)
{
    __shared__ float sU[3], qU[3];
    const int tid = threadIdx.x;
    if (tid < 3) { sU[tid] = 0.f; qU[tid] = 0.f; }
    __syncthreads();

    const int row = blockIdx.x * 256 + tid;
    const int n   = row >> 4;
    const int id  = knn[row];
    ((int*)out)[DO_IDX + row] = id;

    float pr[3];
    #pragma unroll
    for (int i = 0; i < 3; i++) {
        pr[i] = p[(size_t)id * 3 + i] - p[(size_t)n * 3 + i];
        out[DO_PR + (size_t)row * 3 + i] = pr[i];
    }
    #pragma unroll
    for (int i = 0; i < 3; i++) {
        const float u = bp1[i] + pr[0] * Wp1[0 * 3 + i] + pr[1] * Wp1[1 * 3 + i] + pr[2] * Wp1[2 * 3 + i];
        atomicAdd(&sU[i], u);
        atomicAdd(&qU[i], u * u);
    }
    __syncthreads();
    if (tid < 3) {
        ws[WS_PU + (size_t)blockIdx.x * 6 + tid]     = sU[tid];
        ws[WS_PU + (size_t)blockIdx.x * 6 + 3 + tid] = qU[tid];
    }
}

// =====================================================================
// K1b (WMMA): per 16-row tile:
//   h = relu( [p_r | x_gather] @ W1 + b1 )      4 WMMA (p_r part in C-init)
//   q[K] = h[K/16]*h[K%16]  (K=0..255)
//   energy = q @ Wb_flat + bb                   8 WMMA  -> e_in[:,0:16]
// 8 waves/block, 8 tiles/wave. A-fragments are gathered rows of x.
// =====================================================================
__global__ void __launch_bounds__(256)
k1b_energy(const float* __restrict__ x, const int* __restrict__ knn,
           const float* __restrict__ W1, const float* __restrict__ b1,
           const float* __restrict__ Wb, const float* __restrict__ bb,
           const float* __restrict__ out, float* __restrict__ ws)
{
    __shared__ float h_lds[8][16][17];   // per-wave h tile (row-major, padded)
    const int tid = threadIdx.x, lane = tid & 31, wv = tid >> 5;
    const int m = lane & 15, kset = lane >> 4, k0 = kset * 16;
    const int r0 = (lane < 16) ? 0 : 8;

    // B fragments for W1 x-part: rows 3..130 of W1, 4 K-steps of 32
    v16h bW1[4];
    #pragma unroll
    for (int s = 0; s < 4; s++)
        #pragma unroll
        for (int t2 = 0; t2 < 16; t2++)
            bW1[s][t2] = (_Float16)W1[(3 + s * 32 + kset * 8 + (t2 & 7) + (t2 >> 3) * 16) * 16 + m];
    // B fragments for Wb_flat[K][o] = Wb[o*256+K]: 8 K-steps of 32
    v16h bWb[8];
    #pragma unroll
    for (int s = 0; s < 8; s++)
        #pragma unroll
        for (int t2 = 0; t2 < 16; t2++)
            bWb[s][t2] = (_Float16)Wb[m * 256 + (s * 32 + kset * 8 + (t2 & 7) + (t2 >> 3) * 16)];

    const float b1m  = b1[m];
    const float w1p0 = W1[0 * 16 + m], w1p1 = W1[1 * 16 + m], w1p2 = W1[2 * 16 + m];
    const float bbm  = bb[m];
    float* er = ws + WS_EIN;

    for (int t = 0; t < 8; t++) {
        const int tile    = (blockIdx.x * 8 + wv) * 8 + t;
        const int rowBase = tile * 16;

        // ---- GEMM 1: h = relu([p_r|x]@W1 + b1) ----
        v8f c;
        #pragma unroll
        for (int r = 0; r < 8; r++) {
            const int row = rowBase + r0 + r;
            const float* prp = out + DO_PR + (size_t)row * 3;
            c[r] = b1m + prp[0] * w1p0 + prp[1] * w1p1 + prp[2] * w1p2;
        }
        const int id = knn[rowBase + m];            // gathered A-row for this lane
        const float* xr = x + (size_t)id * CIN;
        #pragma unroll
        for (int s = 0; s < 4; s++) {
            const float* xb = xr + s * 32;
            v16h af;
            #pragma unroll
            for (int i = 0; i < 8; i++) {
                af[i]     = (_Float16)xb[kset * 8 + i];
                af[8 + i] = (_Float16)xb[16 + kset * 8 + i];
            }
            c = WMMA_F16(af, bW1[s], c);
        }
        #pragma unroll
        for (int r = 0; r < 8; r++)                 // D: col=m, rows r0+r
            h_lds[wv][r0 + r][m] = fmaxf(c[r], 0.f);
        __syncthreads();

        // ---- GEMM 2: energy = q @ Wb_flat + bb, q[K]=h[K/16]*h[K%16] ----
        const float* hm = &h_lds[wv][m][0];         // A-row m of h
        float lo[8];
        #pragma unroll
        for (int i = 0; i < 8; i++) lo[i] = hm[kset * 8 + i];
        v8f e;
        #pragma unroll
        for (int r = 0; r < 8; r++) e[r] = bbm;
        #pragma unroll
        for (int s = 0; s < 8; s++) {
            const float hi0 = hm[2 * s], hi1 = hm[2 * s + 1];
            v16h aq;
            #pragma unroll
            for (int i = 0; i < 8; i++) {
                aq[i]     = (_Float16)(hi0 * lo[i]);   // K = s*32 + kset*8 + i
                aq[8 + i] = (_Float16)(hi1 * lo[i]);   // K = s*32 + 16 + kset*8 + i
            }
            e = WMMA_F16(aq, bWb[s], e);
        }
        #pragma unroll
        for (int r = 0; r < 8; r++)                 // energy[row][o=m]
            er[(size_t)(rowBase + r0 + r) * 32 + m] = e[r];
        __syncthreads();                            // guard h_lds reuse next tile
    }
}

// =====================================================================
// Deterministic BN finalize: fixed-order reduction over per-block partials.
// =====================================================================
__global__ void kreduce_bn(const float* __restrict__ part, int nblk, int nch,
                           const float* __restrict__ g, const float* __restrict__ be,
                           float* __restrict__ outp)
{
    const int t = threadIdx.x;
    if (t >= nch) return;
    float S = 0.f, Q = 0.f;
    for (int b = 0; b < nblk; b++) {
        S += part[(size_t)b * 2 * nch + t];
        Q += part[(size_t)b * 2 * nch + nch + t];
    }
    const float mean = S / CNTF;
    const float var  = Q / CNTF - mean * mean;
    const float sc   = g[t] * rsqrtf(var + EPSV);
    outp[t]       = sc;
    outp[nch + t] = be[t] - mean * sc;
}

// =====================================================================
// K2: shrink_p -> e_in[:,16:32] via column-folded Wp2 (48 MACs/row).
// =====================================================================
__global__ void __launch_bounds__(256)
k2_shrink(const float* __restrict__ Wp1, const float* __restrict__ bp1,
          const float* __restrict__ Wp2, const float* __restrict__ bp2,
          const float* __restrict__ out, float* __restrict__ ws)
{
    __shared__ float Wp2s[48];
    __shared__ float bp2s[16];
    __shared__ float bn1s[6];
    const int tid = threadIdx.x;
    if (tid < 48) {
        const int j = tid / 16, b = tid % 16;
        float s = 0.f;
        #pragma unroll
        for (int a = 0; a < 8; a++) s += Wp2[j * 128 + a * 16 + b];
        Wp2s[tid] = s;
    } else if (tid < 64) {
        const int b = tid - 48;
        float s = 0.f;
        #pragma unroll
        for (int a = 0; a < 8; a++) s += bp2[a * 16 + b];
        bp2s[b] = s;
    } else if (tid < 70) {
        bn1s[tid - 64] = ws[WS_BN1 + (tid - 64)];
    }
    __syncthreads();

    const int row = blockIdx.x * 256 + tid;
    const float* pr = out + DO_PR + (size_t)row * 3;
    const float pr0 = pr[0], pr1 = pr[1], pr2 = pr[2];
    float t3[3];
    #pragma unroll
    for (int i = 0; i < 3; i++) {
        const float u = bp1[i] + pr0 * Wp1[0 * 3 + i] + pr1 * Wp1[1 * 3 + i] + pr2 * Wp1[2 * 3 + i];
        t3[i] = fmaxf(u * bn1s[i] + bn1s[3 + i], 0.f);
    }
    float* er = ws + WS_EIN + (size_t)row * 32 + 16;
    #pragma unroll
    for (int b = 0; b < 16; b++)
        er[b] = bp2s[b] + t3[0] * Wp2s[b] + t3[1] * Wp2s[16 + b] + t3[2] * Wp2s[32 + b];
}

// =====================================================================
// P2 (WMMA): a = e_in[1M,32] @ W2a[32,128]; BN2a stats only.
// =====================================================================
__global__ void __launch_bounds__(256)
p2_w2a_stats(const float* __restrict__ W2a, float* __restrict__ ws)
{
    __shared__ float sA[128], qA[128];
    const int tid = threadIdx.x, lane = tid & 31, wv = tid >> 5;
    if (tid < 128) { sA[tid] = 0.f; qA[tid] = 0.f; }
    __syncthreads();

    const int colBase = wv * 16, m = lane & 15, kset = lane >> 4;
    v16h bf;
    {
        const int nn = colBase + m, k0 = kset * 16;
        #pragma unroll
        for (int t2 = 0; t2 < 16; t2++) bf[t2] = (_Float16)W2a[(kset * 8 + (t2 & 7) + (t2 >> 3) * 16) * 128 + nn];
    }
    float rs = 0.f, rq = 0.f;
    const float* EIN = ws + WS_EIN;
    for (int t = 0; t < 64; t++) {
        const int rowBase = (blockIdx.x * 64 + t) * 16;
        const float* arow = EIN + (size_t)(rowBase + m) * 32;
        v16h af;
        #pragma unroll
        for (int i = 0; i < 8; i++) {
            af[i]     = (_Float16)arow[kset * 8 + i];
            af[8 + i] = (_Float16)arow[16 + kset * 8 + i];
        }
        v8f c = {};
        c = WMMA_F16(af, bf, c);
        #pragma unroll
        for (int r = 0; r < 8; r++) { const float v = c[r]; rs += v; rq += v * v; }
    }
    atomicAdd(&sA[colBase + m], rs);
    atomicAdd(&qA[colBase + m], rq);
    __syncthreads();
    if (tid < 128) {
        ws[WS_PA + (size_t)blockIdx.x * 256 + tid]       = sA[tid];
        ws[WS_PA + (size_t)blockIdx.x * 256 + 128 + tid] = qA[tid];
    }
}

// =====================================================================
// P3 (WMMA): recompute a, bn2a+relu -> LDS f16; wave0: b_pre = ar@W2b,
// BN2b partial stats.
// =====================================================================
__global__ void __launch_bounds__(256)
p3_w2b(const float* __restrict__ W2a, const float* __restrict__ W2b,
       float* __restrict__ ws)
{
    __shared__ _Float16 ah[16][128];
    __shared__ float sB[16], qB[16];
    __shared__ float bn2a_s[256];
    const int tid = threadIdx.x, lane = tid & 31, wv = tid >> 5;
    if (tid < 16) { sB[tid] = 0.f; qB[tid] = 0.f; }
    bn2a_s[tid] = ws[WS_BN2A + tid];
    __syncthreads();

    const int colBase = wv * 16, m = lane & 15, kset = lane >> 4;
    v16h bfA;
    {
        const int nn = colBase + m, k0 = kset * 16;
        #pragma unroll
        for (int t2 = 0; t2 < 16; t2++) bfA[t2] = (_Float16)W2a[(kset * 8 + (t2 & 7) + (t2 >> 3) * 16) * 128 + nn];
    }
    v16h bfB[4];
    if (wv == 0) {
        const int k0 = kset * 16;
        #pragma unroll
        for (int s = 0; s < 4; s++)
            #pragma unroll
            for (int t2 = 0; t2 < 16; t2++)
                bfB[s][t2] = (_Float16)W2b[(s * 32 + kset * 8 + (t2 & 7) + (t2 >> 3) * 16) * 16 + m];
    }
    const int ch  = colBase + m;
    const float scl = bn2a_s[ch], shf = bn2a_s[128 + ch];
    float rs = 0.f, rq = 0.f;
    const float* EIN  = ws + WS_EIN;
    float*       BPRE = ws + WS_BPRE;
    const int r0 = (lane < 16) ? 0 : 8;

    for (int t = 0; t < 64; t++) {
        const int rowBase = (blockIdx.x * 64 + t) * 16;
        const float* arow = EIN + (size_t)(rowBase + m) * 32;
        v16h af;
        #pragma unroll
        for (int i = 0; i < 8; i++) {
            af[i]     = (_Float16)arow[kset * 8 + i];
            af[8 + i] = (_Float16)arow[16 + kset * 8 + i];
        }
        v8f c = {};
        c = WMMA_F16(af, bfA, c);
        #pragma unroll
        for (int r = 0; r < 8; r++)
            ah[r0 + r][ch] = (_Float16)fmaxf(c[r] * scl + shf, 0.f);
        __syncthreads();

        if (wv == 0) {
            v8f d = {};
            #pragma unroll
            for (int s = 0; s < 4; s++) {
                v16h af2;
                #pragma unroll
                for (int i = 0; i < 8; i++) {
                    af2[i]     = ah[m][s * 32 + kset * 8 + i];
                    af2[8 + i] = ah[m][s * 32 + 16 + kset * 8 + i];
                }
                d = WMMA_F16(af2, bfB[s], d);
            }
            #pragma unroll
            for (int r = 0; r < 8; r++) {
                const float v = d[r];
                BPRE[(size_t)(rowBase + r0 + r) * 16 + m] = v;
                rs += v; rq += v * v;
            }
        }
        __syncthreads();
    }
    if (wv == 0) { atomicAdd(&sB[m], rs); atomicAdd(&qB[m], rq); }
    __syncthreads();
    if (tid < 16) {
        ws[WS_PB + (size_t)blockIdx.x * 32 + tid]      = sB[tid];
        ws[WS_PB + (size_t)blockIdx.x * 32 + 16 + tid] = qB[tid];
    }
}

// =====================================================================
// KY (WMMA): y = x[65536,128] @ W3[128,128] + b3.
// =====================================================================
__global__ void __launch_bounds__(256)
ky_xw3(const float* __restrict__ x, const float* __restrict__ W3,
       const float* __restrict__ b3, float* __restrict__ ws)
{
    const int tid = threadIdx.x, lane = tid & 31, wv = tid >> 5;
    const int rowBase = blockIdx.x * 16;
    const int colBase = wv * 16, m = lane & 15, kset = lane >> 4;
    const int nn = colBase + m, k0 = kset * 16;

    v8f c = {};
    #pragma unroll
    for (int s = 0; s < 4; s++) {
        const int kb = s * 32;
        v16h af, bf;
        const float* xr = x + (size_t)(rowBase + m) * 128 + kb;
        #pragma unroll
        for (int i = 0; i < 8; i++) {
            af[i]     = (_Float16)xr[kset * 8 + i];
            af[8 + i] = (_Float16)xr[16 + kset * 8 + i];
        }
        #pragma unroll
        for (int t2 = 0; t2 < 16; t2++) bf[t2] = (_Float16)W3[(kb + kset * 8 + (t2 & 7) + (t2 >> 3) * 16) * 128 + nn];
        c = WMMA_F16(af, bf, c);
    }
    float* Y = ws + WS_Y;
    const int r0 = (lane < 16) ? 0 : 8;
    const float bias = b3[nn];
    #pragma unroll
    for (int r = 0; r < 8; r++)
        Y[(size_t)(rowBase + r0 + r) * 128 + nn] = c[r] + bias;
}

// =====================================================================
// KF: one wave per point. bn2b+relu(b_pre)@W2c+b2c -> softmax over k ->
// gather y rows, recompute pe from p_r, write x_knn and x_out.
// =====================================================================
__global__ void __launch_bounds__(256)
kf_final(const int* __restrict__ knn,
         const float* __restrict__ Wp1, const float* __restrict__ bp1,
         const float* __restrict__ Wp2, const float* __restrict__ bp2,
         const float* __restrict__ W2c, const float* __restrict__ b2c,
         float* __restrict__ out, float* __restrict__ ws)
{
    __shared__ int   idxs[8][16];
    __shared__ float tv[8][16][3];
    __shared__ float ew[8][16][17];
    __shared__ float bn1s[6], bn2bs[32];
    const int tid = threadIdx.x, lane = tid & 31, wv = tid >> 5;
    if (tid < 6)  bn1s[tid]  = ws[WS_BN1 + tid];
    if (tid < 32) bn2bs[tid] = ws[WS_BN2B + tid];
    __syncthreads();

    const int n = blockIdx.x * 8 + wv;
    const float* Y    = ws + WS_Y;
    const float* BPRE = ws + WS_BPRE;

    if (lane < 16) {
        const int k  = lane;
        const int id = knn[(size_t)n * 16 + k];
        idxs[wv][k] = id;
        __builtin_prefetch(Y + (size_t)id * 128, 0, 3);   // global_prefetch_b8
        const float* pr = out + DO_PR + (size_t)(n * 16 + k) * 3;
        const float pr0 = pr[0], pr1 = pr[1], pr2 = pr[2];
        #pragma unroll
        for (int i = 0; i < 3; i++) {
            const float u = bp1[i] + pr0 * Wp1[0 * 3 + i] + pr1 * Wp1[1 * 3 + i] + pr2 * Wp1[2 * 3 + i];
            tv[wv][k][i] = fmaxf(u * bn1s[i] + bn1s[3 + i], 0.f);
        }
        float br[16];
        const float* bp = BPRE + (size_t)(n * 16 + k) * 16;
        #pragma unroll
        for (int j = 0; j < 16; j++)
            br[j] = fmaxf(bp[j] * bn2bs[j] + bn2bs[16 + j], 0.f);
        #pragma unroll
        for (int c = 0; c < 16; c++) {
            float a = b2c[c];
            #pragma unroll
            for (int j = 0; j < 16; j++) a += br[j] * W2c[j * 16 + c];
            ew[wv][k][c] = a;
        }
    }
    __syncthreads();
    if (lane < 16) {
        const int c = lane;
        float mx = -1e30f;
        #pragma unroll
        for (int k = 0; k < 16; k++) mx = fmaxf(mx, ew[wv][k][c]);
        float e[16], s = 0.f;
        #pragma unroll
        for (int k = 0; k < 16; k++) { e[k] = __expf(ew[wv][k][c] - mx); s += e[k]; }
        const float inv = 1.f / s;
        #pragma unroll
        for (int k = 0; k < 16; k++) ew[wv][k][c] = e[k] * inv;
    }
    __syncthreads();

    float acc[4] = {0.f, 0.f, 0.f, 0.f};
    float* xk = out + DO_XKNN + (size_t)n * (KNN * OUTC);
    for (int k = 0; k < 16; k++) {
        const float* yb = Y + (size_t)idxs[wv][k] * 128;
        const float t0 = tv[wv][k][0], t1 = tv[wv][k][1], t2 = tv[wv][k][2];
        const float wgt = ew[wv][k][lane & 15];
        #pragma unroll
        for (int q = 0; q < 4; q++) {
            const int c = q * 32 + lane;
            const float pe = bp2[c] + t0 * Wp2[c] + t1 * Wp2[128 + c] + t2 * Wp2[256 + c];
            const float v  = (yb[c] + pe) * wgt;
            xk[(size_t)k * 128 + c] = v;
            acc[q] += v;
        }
    }
    #pragma unroll
    for (int q = 0; q < 4; q++)
        out[DO_XOUT + (size_t)n * 128 + q * 32 + lane] = acc[q];
}

// =====================================================================
extern "C" void kernel_launch(void* const* d_in, const int* in_sizes, int n_in,
                              void* d_out, int out_size, void* d_ws, size_t ws_size,
                              hipStream_t stream)
{
    (void)in_sizes; (void)n_in; (void)out_size; (void)ws_size;
    const float* p    = (const float*)d_in[0];
    const float* x    = (const float*)d_in[1];
    const int*   knn  = (const int*)  d_in[2];
    const float* W1   = (const float*)d_in[3];
    const float* b1   = (const float*)d_in[4];
    const float* Wb   = (const float*)d_in[5];
    const float* bb   = (const float*)d_in[6];
    const float* Wp1  = (const float*)d_in[7];
    const float* bp1  = (const float*)d_in[8];
    const float* g1   = (const float*)d_in[9];
    const float* be1  = (const float*)d_in[10];
    const float* Wp2  = (const float*)d_in[11];
    const float* bp2  = (const float*)d_in[12];
    const float* W2a  = (const float*)d_in[13];
    const float* g2a  = (const float*)d_in[14];
    const float* be2a = (const float*)d_in[15];
    const float* W2b  = (const float*)d_in[16];
    const float* g2b  = (const float*)d_in[17];
    const float* be2b = (const float*)d_in[18];
    const float* W2c  = (const float*)d_in[19];
    const float* b2c  = (const float*)d_in[20];
    const float* W3   = (const float*)d_in[21];
    const float* b3   = (const float*)d_in[22];
    float* out = (float*)d_out;
    float* ws  = (float*)d_ws;

    k1a_prep<<<4096, 256, 0, stream>>>(p, knn, Wp1, bp1, out, ws);
    k1b_energy<<<1024, 256, 0, stream>>>(x, knn, W1, b1, Wb, bb, out, ws);
    kreduce_bn<<<1, 32, 0, stream>>>(ws + WS_PU, 4096, 3, g1, be1, ws + WS_BN1);
    k2_shrink<<<4096, 256, 0, stream>>>(Wp1, bp1, Wp2, bp2, out, ws);
    p2_w2a_stats<<<1024, 256, 0, stream>>>(W2a, ws);
    kreduce_bn<<<1, 128, 0, stream>>>(ws + WS_PA, 1024, 128, g2a, be2a, ws + WS_BN2A);
    p3_w2b<<<1024, 256, 0, stream>>>(W2a, W2b, ws);
    kreduce_bn<<<1, 32, 0, stream>>>(ws + WS_PB, 1024, 16, g2b, be2b, ws + WS_BN2B);
    ky_xw3<<<4096, 256, 0, stream>>>(x, W3, b3, ws);
    kf_final<<<8192, 256, 0, stream>>>(knn, Wp1, bp1, Wp2, bp2, W2c, b2c, out, ws);
}
